// ModelGCNConcAfterFrame_84104049590283
// MI455X (gfx1250) — matrix operand order repaired and checked by reading, not compiled, not run on hardware
//
#include <hip/hip_runtime.h>

typedef __bf16 bf16_t;
typedef __attribute__((ext_vector_type(16))) __bf16 v16bf;
typedef __attribute__((ext_vector_type(8)))  float  v8f;

// ---------------------------------------------------------------------------
// Fragment helpers (layouts per CDNA5 ISA 7.12.2, wave32)
// A (16x32 bf16): lane l -> row m=l&15; half h=l>>4; elems 0..7 = K[h*8..h*8+7],
//                 elems 8..15 = K[16+h*8 .. 16+h*8+7]
// B (32x16 bf16): lane l -> col n=l&15; elem s -> same K as A elem s
// C/D (16x16 f32): elem r -> row r+8*h, col lane&15
// ---------------------------------------------------------------------------

__device__ __forceinline__ v16bf zero_frag() {
  v16bf a;
#pragma unroll
  for (int i = 0; i < 16; ++i) a[i] = (bf16_t)0.0f;
  return a;
}

__device__ __forceinline__ v16bf a_frag_from_f32(const float* __restrict__ p) {
  // p points at this lane's first K run (8 floats); second run is at p+16
  const float4 f0 = *reinterpret_cast<const float4*>(p);
  const float4 f1 = *reinterpret_cast<const float4*>(p + 4);
  const float4 f2 = *reinterpret_cast<const float4*>(p + 16);
  const float4 f3 = *reinterpret_cast<const float4*>(p + 20);
  v16bf a;
  a[0]  = (bf16_t)f0.x; a[1]  = (bf16_t)f0.y; a[2]  = (bf16_t)f0.z; a[3]  = (bf16_t)f0.w;
  a[4]  = (bf16_t)f1.x; a[5]  = (bf16_t)f1.y; a[6]  = (bf16_t)f1.z; a[7]  = (bf16_t)f1.w;
  a[8]  = (bf16_t)f2.x; a[9]  = (bf16_t)f2.y; a[10] = (bf16_t)f2.z; a[11] = (bf16_t)f2.w;
  a[12] = (bf16_t)f3.x; a[13] = (bf16_t)f3.y; a[14] = (bf16_t)f3.z; a[15] = (bf16_t)f3.w;
  return a;
}

// ---------------------------------------------------------------------------
// Generic GEMM: C[M,N] = A[M,K](f32) * Bw[K,N](bf16) (+bias) (+relu)
// Optionally writes an additional bf16 copy of the output (downstream WMMA B).
// Block: 256 threads = 8 waves; block tile 128x64; wave tile 16x64.
// Ping-pong software pipeline (K % 64 == 0): set1 loads overlap set0 WMMAs and
// vice versa; no fragment rotation copies.
// FULL=true: M % 128 == 0 and N % 64 == 0 -> no bounds checks anywhere.
// ---------------------------------------------------------------------------
template <bool FULL>
__global__ __launch_bounds__(256)
void gemm_bf16_wmma(const float* __restrict__ A, const bf16_t* __restrict__ Bw,
                    const float* __restrict__ bias, float* __restrict__ C,
                    bf16_t* __restrict__ Cbf, int M, int N, int K, int relu) {
  const int lane = threadIdx.x & 31;
  const int wave = threadIdx.x >> 5;              // 0..7
  const int rowBase = blockIdx.x * 128 + wave * 16;
  const int nTile = blockIdx.y * 64;
  const int m = lane & 15, h = lane >> 4;
  const int arow_i = rowBase + m;
  const bool rvalid = FULL ? true : (arow_i < M);
  const float* arow = A + (size_t)arow_i * K;

  v8f acc[4];
#pragma unroll
  for (int j = 0; j < 4; ++j)
#pragma unroll
    for (int r = 0; r < 8; ++r) acc[j][r] = 0.0f;

  auto load_set = [&](int kk, v16bf& af_, v16bf* bf_) {
    if (rvalid) af_ = a_frag_from_f32(arow + kk + h * 8);
    else        af_ = zero_frag();
    const bf16_t* bk = Bw + nTile + m;
#pragma unroll
    for (int j = 0; j < 4; ++j) {
      if (FULL || (nTile + j * 16) < N) {
#pragma unroll
        for (int s = 0; s < 16; ++s)
          bf_[j][s] = bk[(size_t)(kk + h * 8 + (s & 7) + ((s >> 3) << 4)) * N + j * 16];
      }
      else                              bf_[j] = zero_frag();
    }
  };
  auto mma = [&](const v16bf& af_, const v16bf* bf_) {
#pragma unroll
    for (int j = 0; j < 4; ++j)
      acc[j] = __builtin_amdgcn_wmma_f32_16x16x32_bf16(
          false, af_, false, bf_[j], (short)0, acc[j], false, false);
  };

  v16bf a0, a1, b0[4], b1[4];
  load_set(0, a0, b0);
  for (int kb = 0;;) {
    load_set(kb + 32, a1, b1);
    if (kb + 64 < K)  // warm L2/L0 ahead (global_prefetch_b8)
      __builtin_prefetch(Bw + (size_t)(kb + 64 + lane) * N + nTile, 0, 1);
    mma(a0, b0);
    if (kb + 64 >= K) { mma(a1, b1); break; }
    load_set(kb + 64, a0, b0);
    mma(a1, b1);
    kb += 64;
  }

  // ---- epilogue ----
  const int col0 = lane & 15;
#pragma unroll
  for (int j = 0; j < 4; ++j) {
    const int n0 = nTile + j * 16;
    if (!FULL && n0 >= N) continue;
    const int col = n0 + col0;
    const float bv = bias ? bias[col] : 0.0f;
#pragma unroll
    for (int r = 0; r < 8; ++r) {
      const int orow = rowBase + r + 8 * h;
      if (FULL || orow < M) {
        float v = acc[j][r] + bv;
        if (relu) v = fmaxf(v, 0.0f);
        if (C)   C[(size_t)orow * N + col] = v;
        if (Cbf) Cbf[(size_t)orow * N + col] = (bf16_t)v;
      }
    }
  }
}

// ---------------------------------------------------------------------------
// Batched H[b] = Adj[b](64x64 padded f32) * Xbf[b](nn x N bf16), K padded to 64.
// Block: 128 threads = 4 waves covering M=64; grid (nb, N/64). K fully unrolled.
// ---------------------------------------------------------------------------
__global__ __launch_bounds__(128)
void adj_gemm_wmma(const float* __restrict__ Adj, const bf16_t* __restrict__ Xbf,
                   float* __restrict__ H, int nn, int N) {
  const int b = blockIdx.x;
  const int lane = threadIdx.x & 31;
  const int wave = threadIdx.x >> 5;
  const int nTile = blockIdx.y * 64;
  const int m = lane & 15, h = lane >> 4;
  const float* Ab = Adj + (size_t)b * 4096 + (size_t)(wave * 16 + m) * 64;
  const bf16_t* Bb = Xbf + (size_t)b * nn * N + nTile;

  v8f acc[4];
#pragma unroll
  for (int j = 0; j < 4; ++j)
#pragma unroll
    for (int r = 0; r < 8; ++r) acc[j][r] = 0.0f;

#pragma unroll
  for (int kb = 0; kb < 64; kb += 32) {
    v16bf af = a_frag_from_f32(Ab + kb + h * 8);
    const bf16_t* bk = Bb + m;
#pragma unroll
    for (int j = 0; j < 4; ++j) {
      v16bf bf_;
#pragma unroll
      for (int s = 0; s < 16; ++s) {
        const int k = kb + h * 8 + (s & 7) + ((s >> 3) << 4);
        bf_[s] = (k < nn) ? bk[(size_t)k * N + j * 16] : (bf16_t)0.0f;  // pad K rows beyond nn with zeros
      }
      acc[j] = __builtin_amdgcn_wmma_f32_16x16x32_bf16(
          false, af, false, bf_, (short)0, acc[j], false, false);
    }
  }

  const int col0 = lane & 15;
#pragma unroll
  for (int j = 0; j < 4; ++j) {
    const int col = nTile + j * 16 + col0;
#pragma unroll
    for (int r = 0; r < 8; ++r) {
      const int orow = wave * 16 + r + 8 * h;
      if (orow < nn)
        H[(size_t)(b * nn + orow) * N + col] = acc[j][r];
    }
  }
}

// ---------------------------------------------------------------------------
// adj[b] (64x64 padded) = rownorm( (Q[b] . K[b]^T)^2 ), per-batch, nn nodes
// ---------------------------------------------------------------------------
#define MAXN 50
__global__ __launch_bounds__(256)
void build_adj(const float* __restrict__ Q, const float* __restrict__ Kx,
               float* __restrict__ adj, int nn, int NF) {
  const int b = blockIdx.x;
  const float* Qb = Q + (size_t)b * nn * NF;
  const float* Kb = Kx + (size_t)b * nn * NF;
  __shared__ float sq[MAXN * 64];
  __shared__ float sk[MAXN * 64];
  __shared__ float srow[MAXN];
  const int tid = threadIdx.x;
  const int npairs = nn * nn;

  float acc[10];
#pragma unroll
  for (int i = 0; i < 10; ++i) acc[i] = 0.0f;

  for (int kt = 0; kt < NF; kt += 64) {
    __syncthreads();
    for (int idx = tid; idx < nn * 64; idx += 256) {
      const int r = idx >> 6, c = idx & 63;
      sq[idx] = Qb[(size_t)r * NF + kt + c];
      sk[idx] = Kb[(size_t)r * NF + kt + c];
    }
    __syncthreads();
    int pi = 0;
    for (int p = tid; p < npairs; p += 256, ++pi) {
      const int i = p / nn, j = p - i * nn;
      const float* qi = &sq[i * 64];
      const float* kj = &sk[j * 64];
      float s = acc[pi];
#pragma unroll 8
      for (int c = 0; c < 64; ++c) s = fmaf(qi[c], kj[c], s);
      acc[pi] = s;
    }
  }
  __syncthreads();
  {
    int pi = 0;
    for (int p = tid; p < npairs; p += 256, ++pi) {
      const float d = acc[pi];
      sq[p] = d * d;  // reuse sq buffer: npairs <= 2500 <= MAXN*64
    }
  }
  __syncthreads();
  if (tid < nn) {
    float s = 0.0f;
    for (int j = 0; j < nn; ++j) s += sq[tid * nn + j];
    srow[tid] = 1.0f / fmaxf(s, 1e-12f);
  }
  __syncthreads();
  for (int idx = tid; idx < 64 * 64; idx += 256) {
    const int i = idx >> 6, j = idx & 63;
    float v = 0.0f;
    if (i < nn && j < nn) v = sq[i * nn + j] * srow[i];
    adj[(size_t)b * 4096 + idx] = v;
  }
}

// ---------------------------------------------------------------------------
// LayerNorm(+affine) + ReLU per row of length NF
// ---------------------------------------------------------------------------
__global__ __launch_bounds__(256)
void ln_relu(const float* __restrict__ Hin, const float* __restrict__ g,
             const float* __restrict__ bta, float* __restrict__ Xout, int NF) {
  const int row = blockIdx.x;
  const float* hp = Hin + (size_t)row * NF;
  float* op = Xout + (size_t)row * NF;
  __shared__ float red[2][8];
  __shared__ float stats[2];
  const int tid = threadIdx.x;
  float s = 0.0f, ss = 0.0f;
  for (int c = tid; c < NF; c += 256) {
    const float v = hp[c];
    s += v; ss += v * v;
  }
  const int lane = tid & 31, w = tid >> 5;
#pragma unroll
  for (int off = 16; off > 0; off >>= 1) {
    s += __shfl_xor(s, off, 32);
    ss += __shfl_xor(ss, off, 32);
  }
  if (lane == 0) { red[0][w] = s; red[1][w] = ss; }
  __syncthreads();
  if (tid == 0) {
    float S = 0.0f, SS = 0.0f;
    for (int i = 0; i < 8; ++i) { S += red[0][i]; SS += red[1][i]; }
    const float mu = S / NF;
    const float var = SS / NF - mu * mu;
    stats[0] = mu;
    stats[1] = rsqrtf(var + 1e-5f);
  }
  __syncthreads();
  const float mu = stats[0], rs = stats[1];
  for (int c = tid; c < NF; c += 256) {
    const float v = (hp[c] - mu) * rs * g[c] + bta[c];
    op[c] = fmaxf(v, 0.0f);
  }
}

// ---------------------------------------------------------------------------
// O[b, c] = mean over nn nodes of X[b*nn + i, c]
// ---------------------------------------------------------------------------
__global__ __launch_bounds__(256)
void mean_nodes(const float* __restrict__ X, float* __restrict__ O, int nn, int NF) {
  const int b = blockIdx.x;
  const int c = blockIdx.y * 256 + threadIdx.x;
  if (c >= NF) return;
  float s = 0.0f;
  for (int i = 0; i < nn; ++i) s += X[((size_t)b * nn + i) * NF + c];
  O[(size_t)b * NF + c] = s * (1.0f / (float)nn);
}

// ---------------------------------------------------------------------------
// Weight prep: transpose f32 [R][Cc] -> bf16 [Cc][R];  and plain convert
// ---------------------------------------------------------------------------
__global__ __launch_bounds__(256)
void transpose_to_bf16(const float* __restrict__ W, bf16_t* __restrict__ Wt,
                       int R, int Cc) {
  const int idx = blockIdx.x * 256 + threadIdx.x;
  if (idx >= R * Cc) return;
  const int k = idx / R, n = idx - k * R;
  Wt[idx] = (bf16_t)W[(size_t)n * Cc + k];
}

__global__ __launch_bounds__(256)
void convert_bf16(const float* __restrict__ W, bf16_t* __restrict__ Wb, int total) {
  const int idx = blockIdx.x * 256 + threadIdx.x;
  if (idx < total) Wb[idx] = (bf16_t)W[idx];
}

__global__ __launch_bounds__(256)
void concat_kernel(const float* __restrict__ a, const float* __restrict__ bgl,
                   float* __restrict__ y, int rows, int d) {
  const int idx = blockIdx.x * 256 + threadIdx.x;
  if (idx >= rows * 2 * d) return;
  const int r = idx / (2 * d), c = idx - r * 2 * d;
  y[idx] = (c < d) ? a[(size_t)r * d + c] : bgl[(size_t)r * d + (c - d)];
}

// ---------------------------------------------------------------------------
extern "C" void kernel_launch(void* const* d_in, const int* in_sizes, int n_in,
                              void* d_out, int out_size, void* d_ws, size_t ws_size,
                              hipStream_t stream) {
  (void)in_sizes; (void)n_in; (void)out_size; (void)ws_size;
  const float* feats = (const float*)d_in[0];
  const float* fglob = (const float*)d_in[1];
  const float* wq_w  = (const float*)d_in[3];
  const float* wq_b  = (const float*)d_in[4];
  const float* wk_w  = (const float*)d_in[5];
  const float* wk_b  = (const float*)d_in[6];
  const float* gcn_w = (const float*)d_in[7];
  const float* ln_g  = (const float*)d_in[8];
  const float* ln_b  = (const float*)d_in[9];
  const float* fc1_w = (const float*)d_in[10];
  const float* fc1_b = (const float*)d_in[11];
  const float* fc2_w = (const float*)d_in[12];
  const float* fc2_b = (const float*)d_in[13];

  const int NF = 768, NC = 400;
  const int NB1 = 512, NN1 = 50, M1 = NB1 * NN1;  // 25600
  const int NB2 = 16,  NN2 = 32, M2 = NB2 * NN2;  // 512

  size_t off = 0;
  auto carve = [&](size_t bytes) -> void* {
    void* p = (char*)d_ws + off;
    off += (bytes + 255) & ~(size_t)255;
    return p;
  };
  bf16_t* wqt   = (bf16_t*)carve((size_t)NF * NF * 2);
  bf16_t* wkt   = (bf16_t*)carve((size_t)NF * NF * 2);
  bf16_t* gcnbf = (bf16_t*)carve((size_t)2 * NF * NF * 2);
  bf16_t* fc1t  = (bf16_t*)carve((size_t)2 * NF * NF * 2);      // [1536][768]
  bf16_t* fc2t  = (bf16_t*)carve((size_t)NF * NC * 2);          // [768][400]
  float*  bufA  = (float*)carve((size_t)M1 * NF * 4);           // qx1 / h1
  float*  bufB  = (float*)carve((size_t)M1 * NF * 4);           // kx1 / x1
  bf16_t* xwbf  = (bf16_t*)carve((size_t)M1 * NF * 2);
  float*  adj1  = (float*)carve((size_t)NB1 * 4096 * 4);
  float*  out1  = (float*)carve((size_t)M2 * NF * 4);
  float*  qx2   = (float*)carve((size_t)M2 * NF * 4);
  float*  kx2   = (float*)carve((size_t)M2 * NF * 4);
  bf16_t* xw2bf = (bf16_t*)carve((size_t)M2 * NF * 2);
  float*  adj2  = (float*)carve((size_t)NB2 * 4096 * 4);
  float*  h2    = (float*)carve((size_t)M2 * NF * 4);
  float*  x2    = (float*)carve((size_t)M2 * NF * 4);
  float*  x2out = (float*)carve((size_t)NB2 * NF * 4);
  float*  hfc   = (float*)carve((size_t)NB2 * NF * 4);

  // ---- weight prep (bf16, B-operand layout [K][N]) ----
  transpose_to_bf16<<<(NF * NF + 255) / 256, 256, 0, stream>>>(wq_w, wqt, NF, NF);
  transpose_to_bf16<<<(NF * NF + 255) / 256, 256, 0, stream>>>(wk_w, wkt, NF, NF);
  convert_bf16<<<(2 * NF * NF + 255) / 256, 256, 0, stream>>>(gcn_w, gcnbf, 2 * NF * NF);
  transpose_to_bf16<<<(2 * NF * NF + 255) / 256, 256, 0, stream>>>(fc1_w, fc1t, NF, 2 * NF);
  transpose_to_bf16<<<(NF * NC + 255) / 256, 256, 0, stream>>>(fc2_w, fc2t, NC, NF);

  const dim3 blk128(128), blk256(256);
  const dim3 g1(M1 / 128, NF / 64);   // 200 x 12, full tiles
  const dim3 g2(M2 / 128, NF / 64);   // 4 x 12, full tiles

  // ================= Stage 1: object-level graph (512 batches of 50) =======
  gemm_bf16_wmma<true><<<g1, blk256, 0, stream>>>(feats, wqt, wq_b, bufA, nullptr, M1, NF, NF, 0);
  gemm_bf16_wmma<true><<<g1, blk256, 0, stream>>>(feats, wkt, wk_b, bufB, nullptr, M1, NF, NF, 0);
  build_adj<<<NB1, blk256, 0, stream>>>(bufA, bufB, adj1, NN1, NF);

  const float* x_in = feats;
  for (int l = 0; l < 2; ++l) {
    gemm_bf16_wmma<true><<<g1, blk256, 0, stream>>>(x_in, gcnbf + (size_t)l * NF * NF, nullptr,
                                                    nullptr, xwbf, M1, NF, NF, 0);
    adj_gemm_wmma<<<dim3(NB1, NF / 64), blk128, 0, stream>>>(adj1, xwbf, bufA, NN1, NF);
    ln_relu<<<M1, blk256, 0, stream>>>(bufA, ln_g + l * NF, ln_b + l * NF, bufB, NF);
    x_in = bufB;
  }
  mean_nodes<<<dim3(NB1, NF / 256), blk256, 0, stream>>>(bufB, out1, NN1, NF);

  // ================= Stage 2: frame-level graph (16 batches of 32) =========
  gemm_bf16_wmma<true><<<g2, blk256, 0, stream>>>(out1, wqt, wq_b, qx2, nullptr, M2, NF, NF, 0);
  gemm_bf16_wmma<true><<<g2, blk256, 0, stream>>>(out1, wkt, wk_b, kx2, nullptr, M2, NF, NF, 0);
  build_adj<<<NB2, blk256, 0, stream>>>(qx2, kx2, adj2, NN2, NF);

  const float* x2_in = out1;
  for (int l = 0; l < 2; ++l) {
    gemm_bf16_wmma<true><<<g2, blk256, 0, stream>>>(x2_in, gcnbf + (size_t)l * NF * NF, nullptr,
                                                    nullptr, xw2bf, M2, NF, NF, 0);
    adj_gemm_wmma<<<dim3(NB2, NF / 64), blk128, 0, stream>>>(adj2, xw2bf, h2, NN2, NF);
    ln_relu<<<M2, blk256, 0, stream>>>(h2, ln_g + l * NF, ln_b + l * NF, x2, NF);
    x2_in = x2;
  }
  mean_nodes<<<dim3(NB2, NF / 256), blk256, 0, stream>>>(x2, x2out, NN2, NF);

  // ================= Classifier head =======================================
  float* yout = (float*)d_out + (size_t)NB2 * NC;   // y: [16, 1536], output #2
  concat_kernel<<<(NB2 * 2 * NF + 255) / 256, blk256, 0, stream>>>(x2out, fglob, yout, NB2, NF);
  gemm_bf16_wmma<false><<<dim3(1, NF / 64), blk256, 0, stream>>>(yout, fc1t, fc1_b, hfc, nullptr,
                                                                 NB2, NF, 2 * NF, 1);
  gemm_bf16_wmma<false><<<dim3(1, (NC + 63) / 64), blk256, 0, stream>>>(hfc, fc2t, fc2_b,
                                                                        (float*)d_out, nullptr,
                                                                        NB2, NC, NF, 0);
}
